// Backbone_4793183502914
// MI455X (gfx1250) — matrix operand order repaired and checked by reading, not compiled, not run on hardware
//
#include <hip/hip_runtime.h>

#define B_   8
#define DM   512
#define KNN  16

typedef _Float16 half_t;
typedef __attribute__((ext_vector_type(16))) _Float16 v16h;
typedef __attribute__((ext_vector_type(8)))  _Float16 h8;
typedef __attribute__((ext_vector_type(8)))  float    v8f;

static inline int rup32(int k) { return (k + 31) & ~31; }

// ============================================================================
// WMMA linear on f16 operands, f32 accumulate:
//   C[M x Nout] = act( (A[M x Kp] @ W^T + bias) * bn + be ) (+resid)
// Preconditions: M%16==0, Kp%32==0, Ncomp%16==0; A row stride lda (zero-padded),
// W is [Ncomp x Kp] f16 zero-padded. One wave per 16x64 strip. Output goes to
// Ch (f16) if non-null else Cf (f32); cols in [Nout, ldc) get zero-filled.
// ============================================================================
__global__ __launch_bounds__(32)
void k_linear(const half_t* __restrict__ A, int lda,
              const half_t* __restrict__ W,
              const float* __restrict__ bias, const float* __restrict__ gamma,
              const float* __restrict__ beta,
              const half_t* __restrict__ resid, int ldr,
              float* __restrict__ Cf, half_t* __restrict__ Ch, int ldc,
              int M, int Ncomp, int Nout, int Kp, int relu)
{
    const int lane    = threadIdx.x;        // 0..31
    const int halfSel = lane >> 4;          // 0 | 1
    const int l15     = lane & 15;
    const int tm = blockIdx.y * 16;
    const int tn = blockIdx.x * 64;
    const int rem = Ncomp - tn;
    const int nt  = (rem >= 64) ? 4 : (rem >> 4);   // 1..4, uniform

    const half_t* __restrict__ ap  = A + (size_t)(tm + l15) * lda;
    const half_t* __restrict__ wp0 = W + (size_t)(tn + l15) * Kp;
    const half_t* __restrict__ wp1 = wp0 + (size_t)16 * Kp;
    const half_t* __restrict__ wp2 = wp0 + (size_t)32 * Kp;
    const half_t* __restrict__ wp3 = wp0 + (size_t)48 * Kp;

    v8f acc0 = {}, acc1 = {}, acc2 = {}, acc3 = {};

    if (nt == 4) {
        // Dominant branchless path: batch all 10 b128 loads, then 4 WMMAs.
        for (int k0 = 0; k0 < Kp; k0 += 32) {
            const int ka0 = k0 + halfSel * 8;
            const int kb  = k0 + halfSel * 8;
            h8 alo = *(const h8*)(ap + ka0);          // A halves 0-7
            h8 ahi = *(const h8*)(ap + ka0 + 16);     // A halves 8-15
            h8 b0l = *(const h8*)(wp0 + kb), b0h = *(const h8*)(wp0 + kb + 16);
            h8 b1l = *(const h8*)(wp1 + kb), b1h = *(const h8*)(wp1 + kb + 16);
            h8 b2l = *(const h8*)(wp2 + kb), b2h = *(const h8*)(wp2 + kb + 16);
            h8 b3l = *(const h8*)(wp3 + kb), b3h = *(const h8*)(wp3 + kb + 16);
            v16h af, bf0, bf1, bf2, bf3;
#pragma unroll
            for (int h = 0; h < 8; ++h) {
                af[h]  = alo[h]; af[h + 8]  = ahi[h];
                bf0[h] = b0l[h]; bf0[h + 8] = b0h[h];
                bf1[h] = b1l[h]; bf1[h + 8] = b1h[h];
                bf2[h] = b2l[h]; bf2[h + 8] = b2h[h];
                bf3[h] = b3l[h]; bf3[h + 8] = b3h[h];
            }
            acc0 = __builtin_amdgcn_wmma_f32_16x16x32_f16(false, af, false, bf0, (short)0, acc0, false, false);
            acc1 = __builtin_amdgcn_wmma_f32_16x16x32_f16(false, af, false, bf1, (short)0, acc1, false, false);
            acc2 = __builtin_amdgcn_wmma_f32_16x16x32_f16(false, af, false, bf2, (short)0, acc2, false, false);
            acc3 = __builtin_amdgcn_wmma_f32_16x16x32_f16(false, af, false, bf3, (short)0, acc3, false, false);
        }
    } else {
        for (int k0 = 0; k0 < Kp; k0 += 32) {
            const int ka0 = k0 + halfSel * 8;
            h8 alo = *(const h8*)(ap + ka0);
            h8 ahi = *(const h8*)(ap + ka0 + 16);
            v16h af;
#pragma unroll
            for (int h = 0; h < 8; ++h) { af[h] = alo[h]; af[h + 8] = ahi[h]; }
            const int kb = k0 + halfSel * 8;
#pragma unroll
            for (int t = 0; t < 4; ++t) {
                if (t >= nt) continue;               // uniform branch, EXEC stays full
                const half_t* wpt = (t == 0) ? wp0 : (t == 1) ? wp1 : (t == 2) ? wp2 : wp3;
                h8 blo = *(const h8*)(wpt + kb);
                h8 bhi = *(const h8*)(wpt + kb + 16);
                v16h bf;
#pragma unroll
                for (int h = 0; h < 8; ++h) { bf[h] = blo[h]; bf[h + 8] = bhi[h]; }
                if (t == 0) acc0 = __builtin_amdgcn_wmma_f32_16x16x32_f16(false, af, false, bf, (short)0, acc0, false, false);
                if (t == 1) acc1 = __builtin_amdgcn_wmma_f32_16x16x32_f16(false, af, false, bf, (short)0, acc1, false, false);
                if (t == 2) acc2 = __builtin_amdgcn_wmma_f32_16x16x32_f16(false, af, false, bf, (short)0, acc2, false, false);
                if (t == 3) acc3 = __builtin_amdgcn_wmma_f32_16x16x32_f16(false, af, false, bf, (short)0, acc3, false, false);
            }
        }
    }

    // ---- epilogue ----
#pragma unroll
    for (int t = 0; t < 4; ++t) {
        if (t >= nt) continue;
        const int col = tn + 16 * t + l15;
        const v8f a = (t == 0) ? acc0 : (t == 1) ? acc1 : (t == 2) ? acc2 : acc3;
        if (col < Nout) {
            const float bi = bias  ? bias[col] : 0.0f;
            const float g  = gamma ? gamma[col] * 0.99999500003749968f /* 1/sqrt(1+1e-5) */ : 1.0f;
            const float be = beta  ? beta[col] : 0.0f;
#pragma unroll
            for (int r = 0; r < 8; ++r) {
                const int row = tm + r + halfSel * 8;
                float z = a[r] + bi;
                if (gamma) z = z * g + be;
                if (relu)  z = fmaxf(z, 0.0f);
                if (resid) z += (float)resid[(size_t)row * ldr + col];
                if (Ch) Ch[(size_t)row * ldc + col] = (half_t)z;
                else    Cf[(size_t)row * ldc + col] = z;
            }
        } else if (col < ldc) {                      // zero-fill K-padding columns
#pragma unroll
            for (int r = 0; r < 8; ++r) {
                const int row = tm + r + halfSel * 8;
                if (Ch) Ch[(size_t)row * ldc + col] = (half_t)0.0f;
                else    Cf[(size_t)row * ldc + col] = 0.0f;
            }
        }
    }
}

// Pad/convert a [rows x cols] f32 row-major matrix into [prows x pcols] f16.
__global__ void k_padw_h(const float* __restrict__ src, half_t* __restrict__ dst,
                         int rows, int cols, int prows, int pcols)
{
    int t = blockIdx.x * blockDim.x + threadIdx.x;
    if (t >= prows * pcols) return;
    const int r = t / pcols, c = t - r * pcols;
    dst[t] = (r < rows && c < cols) ? (half_t)src[(size_t)r * cols + c] : (half_t)0.0f;
}

// ============================================================================
// kNN: per query, stable top-16 smallest squared distances (matches top_k(-d)).
// ============================================================================
__global__ void k_knn(const float* __restrict__ Qz, const float* __restrict__ Rz,
                      int* __restrict__ out, int Mq, int Nr)
{
    int t = blockIdx.x * blockDim.x + threadIdx.x;
    if (t >= B_ * Mq) return;
    const int b = t / Mq;
    const float qx = Qz[(size_t)t * 3 + 0];
    const float qy = Qz[(size_t)t * 3 + 1];
    const float qz = Qz[(size_t)t * 3 + 2];
    const float* __restrict__ r = Rz + (size_t)b * Nr * 3;

    float dk[KNN]; int ik[KNN];
#pragma unroll
    for (int j = 0; j < KNN; ++j) { dk[j] = 3.4e38f; ik[j] = 0; }

    for (int n = 0; n < Nr; ++n) {
        const float dx = qx - r[n * 3 + 0];
        const float dy = qy - r[n * 3 + 1];
        const float dz = qz - r[n * 3 + 2];
        const float d = dx * dx + dy * dy + dz * dz;
        if (d < dk[KNN - 1]) {
            dk[KNN - 1] = d; ik[KNN - 1] = n;
#pragma unroll
            for (int s = KNN - 1; s > 0; --s) {
                if (dk[s] < dk[s - 1]) {
                    float td = dk[s]; dk[s] = dk[s - 1]; dk[s - 1] = td;
                    int   ti = ik[s]; ik[s] = ik[s - 1]; ik[s - 1] = ti;
                }
            }
        }
    }
#pragma unroll
    for (int j = 0; j < KNN; ++j) out[(size_t)t * KNN + j] = ik[j];
}

// ============================================================================
// Farthest point sampling (one block per batch, replicates reference scan).
// ============================================================================
__global__ __launch_bounds__(256)
void k_fps(const float* __restrict__ xyz, int N, int npoint, int* __restrict__ fidx)
{
    __shared__ float sd[1024];
    __shared__ float sv[256];
    __shared__ int   si[256];
    __shared__ int   sfar;
    const int b = blockIdx.x, tid = threadIdx.x;
    const float* __restrict__ xb = xyz + (size_t)b * N * 3;

    for (int i = tid; i < N; i += 256) sd[i] = 1e10f;
    if (tid == 0) sfar = 0;
    __syncthreads();

    for (int it = 0; it < npoint; ++it) {
        const int far = sfar;
        if (tid == 0) fidx[b * npoint + it] = far;   // record BEFORE update
        const float cx = xb[far * 3], cy = xb[far * 3 + 1], cz = xb[far * 3 + 2];
        float bv = -1.0f; int bi = 0;
        for (int i = tid; i < N; i += 256) {
            const float dx = xb[i * 3] - cx, dy = xb[i * 3 + 1] - cy, dz = xb[i * 3 + 2] - cz;
            const float nd = fminf(sd[i], dx * dx + dy * dy + dz * dz);
            sd[i] = nd;
            if (nd > bv) { bv = nd; bi = i; }        // strict > : first max wins
        }
        sv[tid] = bv; si[tid] = bi;
        __syncthreads();
        for (int off = 128; off > 0; off >>= 1) {
            if (tid < off) {
                const float v2 = sv[tid + off]; const int i2 = si[tid + off];
                if (v2 > sv[tid] || (v2 == sv[tid] && i2 < si[tid])) { sv[tid] = v2; si[tid] = i2; }
            }
            __syncthreads();
        }
        if (tid == 0) sfar = si[0];
        __syncthreads();
    }
}

// ============================================================================
// Small elementwise / gather kernels
// ============================================================================
__global__ void k_xyz_from_x(const float* __restrict__ x, float* __restrict__ xyz, int total)
{
    int t = blockIdx.x * blockDim.x + threadIdx.x;
    if (t >= total) return;
    xyz[t * 3 + 0] = x[t * 6 + 0];
    xyz[t * 3 + 1] = x[t * 6 + 1];
    xyz[t * 3 + 2] = x[t * 6 + 2];
}

// 6->8 relu ->8 MLP; f16 output padded to stride 32 (zero tail)
__global__ void k_premlp(const float* __restrict__ x, const float* __restrict__ w1,
                         const float* __restrict__ b1, const float* __restrict__ w2,
                         const float* __restrict__ b2, half_t* __restrict__ H, int total)
{
    int t = blockIdx.x * blockDim.x + threadIdx.x;
    if (t >= total) return;
    const float* xp = x + (size_t)t * 6;
    float h1[8];
#pragma unroll
    for (int j = 0; j < 8; ++j) {
        float s = b1[j];
#pragma unroll
        for (int i = 0; i < 6; ++i) s += w1[j * 6 + i] * xp[i];
        h1[j] = fmaxf(s, 0.0f);
    }
    half_t* o = H + (size_t)t * 32;
#pragma unroll
    for (int j = 0; j < 8; ++j) {
        float s = b2[j];
#pragma unroll
        for (int i = 0; i < 8; ++i) s += w2[j * 8 + i] * h1[i];
        o[j] = (half_t)s;
    }
#pragma unroll
    for (int j = 8; j < 32; ++j) o[j] = (half_t)0.0f;
}

// rel vectors -> f16, padded to stride 32 (zero tail)
__global__ void k_rel(const float* __restrict__ xyz, const int* __restrict__ knn,
                      half_t* __restrict__ rel, int N)
{
    int t = blockIdx.x * blockDim.x + threadIdx.x;
    if (t >= N * KNN) return;
    const int n = t / KNN, id = knn[t];
    half_t* o = rel + (size_t)t * 32;
#pragma unroll
    for (int c = 0; c < 3; ++c) o[c] = (half_t)(xyz[n * 3 + c] - xyz[id * 3 + c]);
#pragma unroll
    for (int c = 3; c < 32; ++c) o[c] = (half_t)0.0f;
}

// a0 = q - gather(k,knn) + pos  (gather fused), f16 output for the g1 GEMM
__global__ void k_qkpos(const float* __restrict__ Q, const float* __restrict__ Kf,
                        const int* __restrict__ knn, const float* __restrict__ POS,
                        half_t* __restrict__ A0, int N)
{
    int t = blockIdx.x * blockDim.x + threadIdx.x;
    if (t >= N * KNN * DM) return;
    const int f = t & (DM - 1);
    const int nj = t >> 9;            // / DM
    const int n = nj / KNN;
    const int id = knn[nj];
    A0[t] = (half_t)(Q[(size_t)n * DM + f] - Kf[(size_t)id * DM + f] + POS[t]);
}

// softmax over the 16-neighbor axis, per (n, f), with 1/sqrt(D_MODEL) scaling
__global__ void k_softmax16(float* __restrict__ Aa, int N)
{
    int t = blockIdx.x * blockDim.x + threadIdx.x;
    if (t >= N * DM) return;
    const int n = t >> 9, f = t & (DM - 1);
    const float sc = 0.04419417382415922f;  // 1/sqrt(512)
    const size_t base = (size_t)n * KNN * DM + f;
    float v[KNN], m = -3.4e38f;
#pragma unroll
    for (int j = 0; j < KNN; ++j) { v[j] = Aa[base + (size_t)j * DM] * sc; m = fmaxf(m, v[j]); }
    float s = 0.0f;
#pragma unroll
    for (int j = 0; j < KNN; ++j) { v[j] = expf(v[j] - m); s += v[j]; }
    const float inv = 1.0f / s;
#pragma unroll
    for (int j = 0; j < KNN; ++j) Aa[base + (size_t)j * DM] = v[j] * inv;
}

// res[n,f] = sum_j a[n,j,f]*(gather(v)[n,j,f]+pos[n,j,f]); f16 out for fc2 GEMM
__global__ void k_attnout(const float* __restrict__ Aa, const float* __restrict__ Vf,
                          const int* __restrict__ knn, const float* __restrict__ POS,
                          half_t* __restrict__ RES, int N)
{
    int t = blockIdx.x * blockDim.x + threadIdx.x;
    if (t >= N * DM) return;
    const int n = t >> 9, f = t & (DM - 1);
    const size_t base = (size_t)n * KNN * DM + f;
    float s = 0.0f;
#pragma unroll
    for (int j = 0; j < KNN; ++j) {
        const int id = knn[n * KNN + j];
        s += Aa[base + (size_t)j * DM] * (Vf[(size_t)id * DM + f] + POS[base + (size_t)j * DM]);
    }
    RES[(size_t)n * DM + f] = (half_t)s;
}

__global__ void k_gather3(const float* __restrict__ xyz, const int* __restrict__ fidx,
                          float* __restrict__ out, int np, int N)
{
    int t = blockIdx.x * blockDim.x + threadIdx.x;
    if (t >= B_ * np) return;
    const int b = t / np;
    const int id = fidx[t];
#pragma unroll
    for (int c = 0; c < 3; ++c)
        out[(size_t)t * 3 + c] = xyz[((size_t)b * N + id) * 3 + c];
}

// feat = concat( xyz[knn]-new_xyz, pts[knn] ) -> f16, padded to stride cinp
__global__ void k_feat(const float* __restrict__ xyzin, const float* __restrict__ xyznew,
                       const float* __restrict__ pts, const int* __restrict__ knn,
                       half_t* __restrict__ feat, int np, int Nin, int Cpts, int cinp)
{
    int t = blockIdx.x * blockDim.x + threadIdx.x;
    if (t >= B_ * np * KNN) return;
    const int b = t / (np * KNN);
    const int rem = t - b * np * KNN;
    const int m = rem / KNN;
    const int id = knn[t];
    half_t* o = feat + (size_t)t * cinp;
#pragma unroll
    for (int c = 0; c < 3; ++c)
        o[c] = (half_t)(xyzin[((size_t)b * Nin + id) * 3 + c] - xyznew[((size_t)b * np + m) * 3 + c]);
    const float* p = pts + ((size_t)b * Nin + id) * Cpts;
    for (int c = 0; c < Cpts; ++c) o[3 + c] = (half_t)p[c];
    for (int c = 3 + Cpts; c < cinp; ++c) o[c] = (half_t)0.0f;
}

// max over 16 samples; f16 output padded to stride chp (zero tail)
__global__ void k_maxpool(const float* __restrict__ L, half_t* __restrict__ out,
                          int np, int ch, int chp)
{
    int t = blockIdx.x * blockDim.x + threadIdx.x;
    if (t >= B_ * np * chp) return;
    const int c = t % chp;
    const int bm = t / chp;
    if (c >= ch) { out[t] = (half_t)0.0f; return; }
    float m = -3.4e38f;
#pragma unroll
    for (int j = 0; j < KNN; ++j)
        m = fmaxf(m, L[((size_t)bm * KNN + j) * chp + c]);
    out[t] = (half_t)m;
}

// ============================================================================
// Host orchestration
// ============================================================================
struct TFp { const float *fc1_w,*fc1_b,*fc2_w,*fc2_b,*d1_w,*d1_b,*d2_w,*d2_b,
                         *g1_w,*g1_b,*g2_w,*g2_b,*wq,*wk,*wv; };
struct TDp { const float *w1,*b1,*g1,*be1,*w2,*b2,*g2,*be2; };

extern "C" void kernel_launch(void* const* d_in, const int* in_sizes, int n_in,
                              void* d_out, int out_size, void* d_ws, size_t ws_size,
                              hipStream_t stream)
{
    (void)in_sizes; (void)n_in; (void)out_size; (void)ws_size;
    auto in = [&](int i) { return (const float*)d_in[i]; };

    auto tf_at = [&](int i) {
        TFp t;
        t.fc1_w = in(i+0);  t.fc1_b = in(i+1);  t.fc2_w = in(i+2);  t.fc2_b = in(i+3);
        t.d1_w  = in(i+4);  t.d1_b  = in(i+5);  t.d2_w  = in(i+6);  t.d2_b  = in(i+7);
        t.g1_w  = in(i+8);  t.g1_b  = in(i+9);  t.g2_w  = in(i+10); t.g2_b  = in(i+11);
        t.wq    = in(i+12); t.wk    = in(i+13); t.wv    = in(i+14);
        return t;
    };
    auto td_at = [&](int i) {
        TDp t;
        t.w1 = in(i+0); t.b1 = in(i+1); t.g1 = in(i+2); t.be1 = in(i+3);
        t.w2 = in(i+4); t.b2 = in(i+5); t.g2 = in(i+6); t.be2 = in(i+7);
        return t;
    };

    // ---- workspace layout (byte allocator, 256B aligned) ----
    char* wsb = (char*)d_ws;
    size_t off = 0;
    auto allocB = [&](size_t bytes) { void* p = wsb + off; off += (bytes + 255) & ~(size_t)255; return p; };
    half_t* Hh    = (half_t*)allocB(8ull * 1024 * 32 * 2);    // premlp out, stride 32
    half_t* X1h   = (half_t*)allocB(8ull * 1024 * 512 * 2);
    float*  Qb    = (float*) allocB(8ull * 1024 * 512 * 4);
    float*  Kb    = (float*) allocB(8ull * 1024 * 512 * 4);
    float*  Vb    = (float*) allocB(8ull * 1024 * 512 * 4);
    half_t* RESh  = (half_t*)allocB(8ull * 1024 * 512 * 2);
    float*  BA    = (float*) allocB(1024ull * KNN * 512 * 4); // g2 out / softmax (per batch)
    half_t* BAh   = (half_t*)allocB(1024ull * KNN * 512 * 2); // qkpos out -> g1 A
    half_t* BBh   = (half_t*)allocB(1024ull * KNN * 512 * 2); // d1/g1 out -> d2/g2 A
    float*  BC    = (float*) allocB(1024ull * KNN * 512 * 4); // pos (f32)
    half_t* RELh  = (half_t*)allocB(1024ull * KNN * 32 * 2);
    half_t* FEATh = (half_t*)allocB(2097152ull * 2);
    half_t* L1h   = (half_t*)allocB(2097152ull * 2);
    float*  L2b   = (float*) allocB(2097152ull * 4);
    half_t* TDPh  = (half_t*)allocB(131072ull * 2);
    half_t* WQh   = (half_t*)allocB(262144ull * 2);
    half_t* WKh   = (half_t*)allocB(262144ull * 2);
    half_t* WVh   = (half_t*)allocB(262144ull * 2);
    half_t* WD2h  = (half_t*)allocB(262144ull * 2);
    half_t* WG1h  = (half_t*)allocB(262144ull * 2);
    half_t* WG2h  = (half_t*)allocB(262144ull * 2);
    half_t* WFC1h = (half_t*)allocB(65536ull * 2);            // 512 x sA
    half_t* WFC2h = (half_t*)allocB(65536ull * 2);            // Ncomp2 x 512
    half_t* WD1h  = (half_t*)allocB(16384ull * 2);            // 512 x 32
    half_t* WT1h  = (half_t*)allocB(16384ull * 2);            // chp x cinp
    half_t* WT2h  = (half_t*)allocB(16384ull * 2);            // chp x chp
    int* knn  = (int*)allocB(8ull * 1024 * KNN * 4);
    int* fidx = (int*)allocB(8ull * 512 * 4);

    auto gb = [](long n) { return (unsigned)((n + 255) / 256); };
    auto lin = [&](const half_t* A, int lda, const half_t* W,
                   const float* bias, const float* g, const float* be,
                   const half_t* r, int ldr, float* Cf, half_t* Ch, int ldc,
                   int M, int Ncomp, int Nout, int Kp, int relu) {
        dim3 grid((Ncomp + 63) / 64, M / 16);
        k_linear<<<grid, 32, 0, stream>>>(A, lda, W, bias, g, be, r, ldr,
                                          Cf, Ch, ldc, M, Ncomp, Nout, Kp, relu);
    };
    auto padw = [&](const float* src, half_t* dst, int rows, int cols, int prows, int pcols) {
        k_padw_h<<<gb((long)prows * pcols), 256, 0, stream>>>(src, dst, rows, cols, prows, pcols);
    };

    // ---- point transformer block (fin f16, row stride sA, Kp = sA) ----
    auto run_tf = [&](const TFp& p, const float* xyz, const half_t* fin, int sA,
                      float* fout, int N, int C) {
        k_knn<<<gb((long)B_ * N), 256, 0, stream>>>(xyz, xyz, knn, N, N);
        padw(p.fc1_w, WFC1h, DM, C, DM, sA);
        padw(p.wq,   WQh,  DM, DM, DM, DM);
        padw(p.wk,   WKh,  DM, DM, DM, DM);
        padw(p.wv,   WVh,  DM, DM, DM, DM);
        padw(p.d1_w, WD1h, DM, 3,  DM, 32);
        padw(p.d2_w, WD2h, DM, DM, DM, DM);
        padw(p.g1_w, WG1h, DM, DM, DM, DM);
        padw(p.g2_w, WG2h, DM, DM, DM, DM);
        const int Ncomp2 = (C < 16) ? 16 : C;      // fc2 compute width
        padw(p.fc2_w, WFC2h, C, DM, Ncomp2, DM);
        lin(fin, sA, WFC1h, p.fc1_b, 0, 0, 0, 0, 0, X1h, DM, B_ * N, DM, DM, sA, 0);
        lin(X1h, DM, WQh, 0, 0, 0, 0, 0, Qb, 0, DM, B_ * N, DM, DM, DM, 0);
        lin(X1h, DM, WKh, 0, 0, 0, 0, 0, Kb, 0, DM, B_ * N, DM, DM, DM, 0);
        lin(X1h, DM, WVh, 0, 0, 0, 0, 0, Vb, 0, DM, B_ * N, DM, DM, DM, 0);
        for (int b = 0; b < B_; ++b) {
            const float* xb = xyz + (size_t)b * N * 3;
            const int*   kb = knn + (size_t)b * N * KNN;
            k_rel<<<gb((long)N * KNN), 256, 0, stream>>>(xb, kb, RELh, N);
            lin(RELh, 32, WD1h, p.d1_b, 0, 0, 0, 0, 0, BBh, DM, N * KNN, DM, DM, 32, 1);
            lin(BBh, DM, WD2h, p.d2_b, 0, 0, 0, 0, BC, 0, DM, N * KNN, DM, DM, DM, 0);
            k_qkpos<<<gb((long)N * KNN * DM), 256, 0, stream>>>(
                Qb + (size_t)b * N * DM, Kb + (size_t)b * N * DM, kb, BC, BAh, N);
            lin(BAh, DM, WG1h, p.g1_b, 0, 0, 0, 0, 0, BBh, DM, N * KNN, DM, DM, DM, 1);
            lin(BBh, DM, WG2h, p.g2_b, 0, 0, 0, 0, BA, 0, DM, N * KNN, DM, DM, DM, 0);
            k_softmax16<<<gb((long)N * DM), 256, 0, stream>>>(BA, N);
            k_attnout<<<gb((long)N * DM), 256, 0, stream>>>(
                BA, Vb + (size_t)b * N * DM, kb, BC, RESh + (size_t)b * N * DM, N);
        }
        lin(RESh, DM, WFC2h, p.fc2_b, 0, 0, fin, sA, fout, 0, C, B_ * N, Ncomp2, C, DM, 0);
    };

    // ---- transition down ----
    auto run_td = [&](const TDp& p, const float* xyzin, const float* ptsin,
                      int Nin, int Cpts, int np, int ch,
                      float* xyzout, half_t* ptsout /* stride chp */, int chp) {
        const int cin = 3 + Cpts;
        const int cinp = rup32(cin);
        k_fps<<<B_, 256, 0, stream>>>(xyzin, Nin, np, fidx);
        k_gather3<<<gb((long)B_ * np), 256, 0, stream>>>(xyzin, fidx, xyzout, np, Nin);
        k_knn<<<gb((long)B_ * np), 256, 0, stream>>>(xyzout, xyzin, knn, np, Nin);
        k_feat<<<gb((long)B_ * np * KNN), 256, 0, stream>>>(
            xyzin, xyzout, ptsin, knn, FEATh, np, Nin, Cpts, cinp);
        padw(p.w1, WT1h, ch, cin, chp, cinp);
        padw(p.w2, WT2h, ch, ch, chp, chp);
        const int M = B_ * np * KNN;
        lin(FEATh, cinp, WT1h, p.b1, p.g1, p.be1, 0, 0, 0, L1h, chp, M, chp, ch, cinp, 1);
        lin(L1h,   chp,  WT2h, p.b2, p.g2, p.be2, 0, 0, L2b, 0, chp, M, chp, ch, chp, 1);
        k_maxpool<<<gb((long)B_ * np * chp), 256, 0, stream>>>(L2b, ptsout, np, ch, chp);
    };

    // ---- d_out layout (floats): pts_final, then (xyz_i, pts_i) per level ----
    float* out = (float*)d_out;
    const size_t o_final = 0;
    const size_t o_xyz0 = 65536,            o_pts0 = o_xyz0 + 24576;
    const size_t o_xyz1 = o_pts0 + 65536,   o_pts1 = o_xyz1 + 12288;
    const size_t o_xyz2 = o_pts1 + 65536,   o_pts2 = o_xyz2 + 6144;
    const size_t o_xyz3 = o_pts2 + 65536,   o_pts3 = o_xyz3 + 3072;
    const size_t o_xyz4 = o_pts3 + 65536,   o_pts4 = o_xyz4 + 1536;
    const size_t oxyz[5] = { o_xyz0, o_xyz1, o_xyz2, o_xyz3, o_xyz4 };
    const size_t opts[5] = { o_pts0, o_pts1, o_pts2, o_pts3, o_pts4 };

    const float* x = in(0);
    k_xyz_from_x<<<gb(8192), 256, 0, stream>>>(x, out + o_xyz0, 8192);
    k_premlp<<<gb(8192), 256, 0, stream>>>(x, in(1), in(2), in(3), in(4), Hh, 8192);

    run_tf(tf_at(5), out + o_xyz0, Hh, 32, out + o_pts0, 1024, 8);

    int Ncur = 1024, Ccur = 8;
    for (int blk = 0; blk < 4; ++blk) {
        const int np = Ncur / 2;
        const int ch = 16 << blk;
        const int chp = rup32(ch);
        const TDp td = td_at(20 + 23 * blk);
        const TFp tf = tf_at(28 + 23 * blk);
        run_td(td, out + oxyz[blk], out + opts[blk], Ncur, Ccur, np, ch,
               out + oxyz[blk + 1], TDPh, chp);
        run_tf(tf, out + oxyz[blk + 1], TDPh, chp, out + opts[blk + 1], np, ch);
        Ncur = np; Ccur = ch;
    }

    hipMemcpyAsync(out + o_final, out + o_pts4, 65536 * sizeof(float),
                   hipMemcpyDeviceToDevice, stream);
}
